// Solver_47218870453037
// MI455X (gfx1250) — compile-verified, matrix operand order checked by reading, not run on hardware
//
#include <hip/hip_runtime.h>
#include <hip/hip_fp16.h>
#include <math.h>

typedef _Float16 v16h __attribute__((ext_vector_type(16)));
typedef float    v8f  __attribute__((ext_vector_type(8)));

#define HID_DIM 64
#define NU_CONST 0.01f
#define WAVES_PER_BLOCK 4
#define PERSISTENT_BLOCKS 512

// Branch-free tanh: tanh(x) = 1 - 2/(exp2(2*log2(e)*x) + 1).
// Hardware v_exp_f32 covers the full range (saturates to +-1), no branches.
__device__ __forceinline__ float fast_tanh(float x)
{
    float e = __builtin_amdgcn_exp2f(x * 2.8853900817779268f); // 2/ln(2)
    float r = __builtin_amdgcn_rcpf(e + 1.0f);
    return __builtin_fmaf(-2.0f, r, 1.0f);
}

__device__ __forceinline__ float fast_rcp(float x) { return __builtin_amdgcn_rcpf(x); }

// Persistent edge-MLP: each wave grid-strides over 16-edge tiles of one channel.
// Weights (4 f16 B-tiles of W1, b1, W2 columns) are loaded ONCE per wave into
// registers. Per tile: gather (lanes 0-15, one edge row each), 4x
// v_wmma_f32_16x16x32_f16 (hidden tiles), fast-tanh + W2 FMAs, split butterfly
// over the 16-lane hidden dim, atomicAdd segment-sum over dst.
__global__ __launch_bounds__(128)
void edge_mlp_kernel(const float* __restrict__ S, int sstride, int norm_input,
                     const float* __restrict__ degrees,
                     const int*   __restrict__ edge_index,   // (2, E) row-major
                     const float* __restrict__ edge_attr,    // (E, 2)
                     const float* __restrict__ W1,           // (6, 64)
                     const float* __restrict__ b1,           // (64)
                     const float* __restrict__ W2,           // (64, 2)
                     const float* __restrict__ b2,           // (2)
                     float* __restrict__ agg, int aggStride,
                     int E)
{
    const int  lane  = threadIdx.x & 31;
    const int  wave  = threadIdx.x >> 5;
    const int  c     = blockIdx.y;
    const int  col16 = lane & 15;
    const bool lo    = lane < 16;

    // ---- Hoisted weights (loop-invariant, live in VGPRs) ----
    // B tile t (16-bit B 32x16 layout): lanes 0-15 hold K=0..7,16..23 (2/VGPR) at
    // column = lane; lanes 16-31 (K=8..15/24..31) stay zero (K padding).
    v16h  bt[4];
    float bias[4], w20[4], w21[4];
    #pragma unroll
    for (int t = 0; t < 4; ++t) {
        const int col = t * 16 + col16;
        v16h b = {};
        if (lo) {
            #pragma unroll
            for (int k = 0; k < 6; ++k)
                b[k] = (_Float16)W1[k * HID_DIM + col];
        }
        bt[t]   = b;
        bias[t] = b1[col];
        w20[t]  = W2[col * 2 + 0];
        w21[t]  = W2[col * 2 + 1];
    }
    const float b2j = (lane & 1) ? b2[1] : b2[0];

    // A matrix (16-bit A 16x32 layout): lanes 0-15 hold row M=lane, K=0..7 in
    // a[0..7] (features in K=0..5); lanes 16-31 hold K=8..15/24..31 -> zero.
    // Padding slots are zeroed once; only a[0..5] of lanes 0-15 change per tile.
    v16h a = {};

    const int ntiles = (E + 15) >> 4;
    const int nwaves = gridDim.x * WAVES_PER_BLOCK;

    for (int tile = blockIdx.x * WAVES_PER_BLOCK + wave; tile < ntiles; tile += nwaves) {
        const int ebase = tile << 4;

        // ---- Gather: lane L (<16) builds edge row ebase+L (clamped at tail;
        // a garbage A row only affects its own D row, which the write guard drops).
        int dstv = 0;
        if (lo) {
            int ge = ebase + lane;
            if (ge > E - 1) ge = E - 1;
            const int src = edge_index[ge];
            const int dst = edge_index[E + ge];
            dstv = dst;
            float sd = S[(size_t)dst * sstride + c];
            float ss = S[(size_t)src * sstride + c];
            const float dd = degrees[dst];
            const float ds = degrees[src];
            if (norm_input) {            // fused agg/deg normalization (pass 2)
                sd *= fast_rcp(dd);
                ss *= fast_rcp(ds);
            }
            a[0] = (_Float16)sd;
            a[1] = (_Float16)dd;
            a[2] = (_Float16)ss;
            a[3] = (_Float16)ds;
            a[4] = (_Float16)edge_attr[(size_t)ge * 2 + 0];
            a[5] = (_Float16)edge_attr[(size_t)ge * 2 + 1];
        }

        // Layer-2 partials: v[2r+j] = sum over this lane's hidden cols of
        // tanh(preact[row r]) * W2[col][j].
        float v[16];
        #pragma unroll
        for (int i = 0; i < 16; ++i) v[i] = 0.f;

        #pragma unroll
        for (int t = 0; t < 4; ++t) {
            v8f acc = {};
            acc = __builtin_amdgcn_wmma_f32_16x16x32_f16(
                /*neg_a=*/false, a, /*neg_b=*/false, bt[t],
                /*c_mod=*/(short)0, acc, /*reuse_a=*/false, /*reuse_b=*/false);
            // acc[r] -> (edge row M = r + 8*(lane>=16), hidden col = 16t+col16)
            #pragma unroll
            for (int r = 0; r < 8; ++r) {
                const float h = fast_tanh(acc[r] + bias[t]);
                v[2 * r + 0] = __builtin_fmaf(h, w20[t], v[2 * r + 0]);
                v[2 * r + 1] = __builtin_fmaf(h, w21[t], v[2 * r + 1]);
            }
        }

        // Split butterfly over the 16 lanes of each half (hidden dim): at stage
        // b, keep the partials whose index bit b matches the lane bit b.
        // Live values 16->8->4->2->1; lane L ends holding index k = L&15.
        #pragma unroll
        for (int bstep = 0; bstep < 4; ++bstep) {
            const int  m   = 1 << bstep;
            const bool bit = (lane & m) != 0;
            #pragma unroll
            for (int i = 0; i < (8 >> bstep); ++i) {
                const float vl = v[2 * i];
                const float vh = v[2 * i + 1];
                const float mine   = bit ? vh : vl;
                const float theirs = bit ? vl : vh;
                v[i] = mine + __shfl_xor(theirs, m, 32);
            }
        }
        // v[0] = sum for (row r = col16>>1, output j = lane&1).

        const int r = col16 >> 1;
        const int e = ebase + r + (lo ? 0 : 8);
        // dst index for this row lives in lane (r + 8*hi) of the gather phase.
        const int d = __shfl(dstv, r + (lo ? 0 : 8), 32);
        if (e < E)
            atomicAdd(&agg[(size_t)d * aggStride + c * 2 + (lane & 1)], v[0] + b2j);
    }
}

// Pointwise residual assembly; normalizes agg1/agg2 by degree on the fly.
__global__ void combine_kernel(const float* __restrict__ fields,
                               const float* __restrict__ agg1,  // (N,6) raw sums
                               const float* __restrict__ agg2,  // (N,8) raw sums
                               const float* __restrict__ degrees,
                               float* __restrict__ out, int N)
{
    int n = blockIdx.x * blockDim.x + threadIdx.x;
    if (n >= N) return;
    const float rdeg = fast_rcp(degrees[n]);
    const float u = fields[(size_t)n * 3 + 0];
    const float v = fields[(size_t)n * 3 + 1];
    float g1[6], g2[8];
    #pragma unroll
    for (int k = 0; k < 6; ++k) g1[k] = agg1[(size_t)n * 6 + k] * rdeg;
    #pragma unroll
    for (int k = 0; k < 8; ++k) g2[k] = agg2[(size_t)n * 8 + k] * rdeg;
    // g1: gu_x,gu_y,gv_x,gv_y,gp_x,gp_y ; g2: gux_x,gux_y,guy_x,guy_y,gvx_x,gvx_y,gvy_x,gvy_y
    const float lap_u = g2[0] + g2[3];
    const float lap_v = g2[4] + g2[7];
    const float continuity = g1[0] + g1[3];
    const float mom_x = u * g1[0] + v * g1[1] + g1[4] - NU_CONST * lap_u;
    const float mom_y = u * g1[2] + v * g1[3] + g1[5] - NU_CONST * lap_v;
    out[(size_t)n * 3 + 0] = continuity;
    out[(size_t)n * 3 + 1] = mom_x;
    out[(size_t)n * 3 + 2] = mom_y;
}

extern "C" void kernel_launch(void* const* d_in, const int* in_sizes, int n_in,
                              void* d_out, int out_size, void* d_ws, size_t ws_size,
                              hipStream_t stream)
{
    (void)n_in; (void)out_size; (void)ws_size;
    const float* fields    = (const float*)d_in[0];
    const float* degrees   = (const float*)d_in[1];
    const float* edge_attr = (const float*)d_in[2];
    const float* W1        = (const float*)d_in[3];
    const float* b1        = (const float*)d_in[4];
    const float* W2        = (const float*)d_in[5];
    const float* b2        = (const float*)d_in[6];
    const int*   edge_idx  = (const int*)d_in[7];
    float* out = (float*)d_out;

    const int N = in_sizes[1];          // degrees: one per node
    const int E = in_sizes[2] / 2;      // edge_attr: (E,2)

    float* agg1 = (float*)d_ws;                 // N*6 raw sums
    float* agg2 = agg1 + (size_t)N * 6;         // N*8 raw sums

    hipMemsetAsync(agg1, 0, (size_t)N * 6 * sizeof(float), stream);
    hipMemsetAsync(agg2, 0, (size_t)N * 8 * sizeof(float), stream);

    const int ntiles = (E + 15) / 16;
    int gx = (ntiles + WAVES_PER_BLOCK - 1) / WAVES_PER_BLOCK;
    if (gx > PERSISTENT_BLOCKS) gx = PERSISTENT_BLOCKS;

    // Pass 1: g(u), g(v), g(p)  (channels of fields, stride 3)
    edge_mlp_kernel<<<dim3(gx, 3), 32 * WAVES_PER_BLOCK, 0, stream>>>(
        fields, 3, /*norm_input=*/0, degrees, edge_idx, edge_attr,
        W1, b1, W2, b2, agg1, 6, E);

    // Pass 2: g(grad_u_x), g(grad_u_y), g(grad_v_x), g(grad_v_y)
    // (channels 0..3 of raw agg1, normalized by degree inside the gather)
    edge_mlp_kernel<<<dim3(gx, 4), 32 * WAVES_PER_BLOCK, 0, stream>>>(
        agg1, 6, /*norm_input=*/1, degrees, edge_idx, edge_attr,
        W1, b1, W2, b2, agg2, 8, E);

    // Pass 3: residual assembly with fused normalization
    combine_kernel<<<(N + 255) / 256, 256, 0, stream>>>(
        fields, agg1, agg2, degrees, out, N);
}
